// ProductQuantizer_22213570855029
// MI455X (gfx1250) — matrix operand order repaired and checked by reading, not compiled, not run on hardware
//
#include <hip/hip_runtime.h>
#include <hip/hip_bf16.h>
#include <math.h>

typedef __attribute__((ext_vector_type(16))) _Float16 v16h;
typedef __attribute__((ext_vector_type(8)))  float    v8f;

// Problem constants (from reference setup_inputs)
static constexpr int N_B  = 32;                 // batch
static constexpr int C_F  = 512;                // channels
static constexpr int T_L  = 2048;               // time
static constexpr int NG   = 4;                  // groups
static constexpr int NK   = 1024;               // codes per group
static constexpr int DG   = 128;                // dims per group
static constexpr int NTOK = N_B * T_L;          // 65536 tokens
static constexpr int NEL  = N_B * C_F * T_L;    // 33554432 output elems
static constexpr float BETA = 0.25f;

// LDS staging: B tile = 16 codes x 128 dims f16, padded to 136 halfs/row so
// the WMMA B-operand ds_load_b128 pattern hits all 64 banks (stride 272B ->
// lane m lands on banks 4m..4m+3).
static constexpr int ROWPAD = 136;              // halfs per padded LDS row
static constexpr int TILEH  = 16 * ROWPAD;      // halfs per tile buffer (2176)

// ---------------------------------------------------------------------------
// Kernel 1: f32 codebook -> f16 copy + per-code squared norms; zero accums.
// ---------------------------------------------------------------------------
__global__ void pq_prep_kernel(const float* __restrict__ cb,
                               _Float16* __restrict__ cb16,
                               float* __restrict__ cnorm,
                               unsigned* __restrict__ counts,
                               float* __restrict__ lossSum) {
  int gk = blockIdx.x * blockDim.x + threadIdx.x;   // 0..4095
  if (gk >= NG * NK) return;
  counts[gk] = 0u;
  if (gk == 0) *lossSum = 0.0f;
  const float* s = cb + (size_t)gk * DG;
  _Float16*    d = cb16 + (size_t)gk * DG;
  float acc = 0.0f;
  #pragma unroll 4
  for (int i = 0; i < DG; ++i) {
    float v = s[i];
    acc += v * v;
    d[i] = (_Float16)v;
  }
  cnorm[gk] = acc;
}

// ---------------------------------------------------------------------------
// Kernel 2: fused distance-GEMM + argmin.
// Block = 8 waves, all on the SAME group; each wave owns 32 tokens
// (two 16-token A tiles in registers). B tiles (16 codes x 128 dims) are
// double-buffered in LDS via global_load_async_to_lds_b128 and shared by all
// 8 waves. dist = ||c||^2 - 2*(x.c)  (||x||^2 is constant per token).
// k-loop manually unrolled by 2 so both buffer bases are static.
// ---------------------------------------------------------------------------
__global__ void __launch_bounds__(256)
pq_argmin_kernel(const float* __restrict__ x,
                 const _Float16* __restrict__ cb16,
                 const float* __restrict__ cnorm,
                 unsigned* __restrict__ codeIdx,
                 unsigned* __restrict__ counts) {
  __shared__ __attribute__((aligned(32))) _Float16 smem[2 * TILEH];  // 8704 B

  int tid  = threadIdx.x;
  int lane = tid & 31;
  int w    = tid >> 5;                 // wave in block 0..7
  int g    = blockIdx.x & 3;
  int tb   = blockIdx.x >> 2;          // 0..255 token-block
  int m    = lane & 15;
  int hi   = lane >> 4;                // 0: lanes 0-15, 1: lanes 16-31
  int dHi  = hi * 16;

  int tokBase = tb * 256 + w * 32;     // 32 tokens per wave

  // ---- A operands: two 16-token tiles, 16-bit A 16x32, contiguous k order ----------
  // elements 0..15 -> K = dHi + e (contiguous, dHi = 16*hi)
  v16h a[2][4];
  #pragma unroll
  for (int tile = 0; tile < 2; ++tile) {
    int token = tokBase + tile * 16 + m;
    int ni = token >> 11;              // / T_L
    int t  = token & (T_L - 1);
    const float* xb = x + ((size_t)ni * C_F + (size_t)g * DG) * T_L + t;
    #pragma unroll
    for (int cd = 0; cd < 4; ++cd) {
      int D = cd * 32;
      #pragma unroll
      for (int e = 0; e < 8; ++e)
        a[tile][cd][e] = (_Float16)xb[(size_t)(D + dHi + e) * T_L];
      #pragma unroll
      for (int e = 0; e < 8; ++e)
        a[tile][cd][8 + e] = (_Float16)xb[(size_t)(D + dHi + 8 + e) * T_L];
    }
  }

  // ---- async staging: thread tid copies 16B chunk (row=tid>>4, chunk=tid&15)
  int srow = tid >> 4, schunk = tid & 15;
  const char* gsrc0 =
      (const char*)(cb16 + (size_t)g * NK * DG) + srow * 256 + schunk * 16;
  // Flat shared pointers carry the LDS byte offset in their low 32 bits.
  unsigned lds0 = (unsigned)(size_t)(smem + srow * ROWPAD + schunk * 8);
  unsigned lds1 = (unsigned)(size_t)(smem + TILEH + srow * ROWPAD + schunk * 8);

  // prologue: stage tile 0 into buffer 0
  asm volatile("global_load_async_to_lds_b128 %0, %1, off"
               :: "v"(lds0), "v"((unsigned long long)(size_t)gsrc0)
               : "memory");
  asm volatile("s_wait_asynccnt 0x0" ::: "memory");
  __syncthreads();

  const _Float16* bbL = smem + m * ROWPAD + hi * 16;   // + parity*TILEH + cd*32
  const float*    cn  = cnorm + g * NK + m;

  float bestV[2][8];
  int   bestI[2][8];
  #pragma unroll
  for (int tile = 0; tile < 2; ++tile)
    #pragma unroll
    for (int r = 0; r < 8; ++r) { bestV[tile][r] = 3.4e38f; bestI[tile][r] = 0; }

  // one k-tile step: 8 WMMAs (two accumulator chains) + argmin update
  auto step = [&](const _Float16* bp, int kt) {
    v8f c0 = {}, c1 = {};
    #pragma unroll
    for (int cd = 0; cd < 4; ++cd) {
      v16h b = *(const v16h*)(bp + cd * 32);
      c0 = __builtin_amdgcn_wmma_f32_16x16x32_f16(
          false, a[0][cd], false, b, (short)0, c0, false, false);
      c1 = __builtin_amdgcn_wmma_f32_16x16x32_f16(
          false, a[1][cd], false, b, (short)0, c1, false, false);
    }
    float cnk   = cn[kt * 16];         // ||code||^2 for this lane's column
    int   kcode = kt * 16 + m;
    #pragma unroll
    for (int r = 0; r < 8; ++r) {
      float d0 = __builtin_fmaf(-2.0f, c0[r], cnk);
      if (d0 < bestV[0][r]) { bestV[0][r] = d0; bestI[0][r] = kcode; }
      float d1 = __builtin_fmaf(-2.0f, c1[r], cnk);
      if (d1 < bestV[1][r]) { bestV[1][r] = d1; bestI[1][r] = kcode; }
    }
  };

  #pragma unroll 1
  for (int kt = 0; kt < NK / 16; kt += 2) {
    // ---- phase 0: compute from buffer 0, prefetch tile kt+1 -> buffer 1 ---
    {
      const char* src = gsrc0 + (size_t)(kt + 1) * (16 * DG * 2);  // kt+1<=63
      asm volatile("global_load_async_to_lds_b128 %0, %1, off"
                   :: "v"(lds1), "v"((unsigned long long)(size_t)src)
                   : "memory");
      step(bbL, kt);
      asm volatile("s_wait_asynccnt 0x0" ::: "memory");
      __syncthreads();
    }
    // ---- phase 1: compute from buffer 1, prefetch tile kt+2 -> buffer 0 ---
    {
      if (kt + 2 < NK / 16) {
        const char* src = gsrc0 + (size_t)(kt + 2) * (16 * DG * 2);
        asm volatile("global_load_async_to_lds_b128 %0, %1, off"
                     :: "v"(lds0), "v"((unsigned long long)(size_t)src)
                     : "memory");
      }
      step(bbL + TILEH, kt + 1);
      asm volatile("s_wait_asynccnt 0x0" ::: "memory");
      __syncthreads();
    }
  }

  // ---- cross-lane argmin within each 16-lane half (C rows 0-7 / 8-15) -----
  #pragma unroll
  for (int tile = 0; tile < 2; ++tile) {
    #pragma unroll
    for (int r = 0; r < 8; ++r) {
      #pragma unroll
      for (int off = 1; off < 16; off <<= 1) {
        float ov = __shfl_xor(bestV[tile][r], off, 32);
        int   oi = __shfl_xor(bestI[tile][r], off, 32);
        if (ov < bestV[tile][r] ||
            (ov == bestV[tile][r] && oi < bestI[tile][r])) {
          bestV[tile][r] = ov; bestI[tile][r] = oi;
        }
      }
    }
    if (m == 0) {  // lane 0 holds rows 0-7, lane 16 holds rows 8-15
      int rowBase = tokBase + tile * 16 + hi * 8;
      #pragma unroll
      for (int r = 0; r < 8; ++r) {
        unsigned idx = (unsigned)bestI[tile][r];
        codeIdx[(size_t)(rowBase + r) * NG + g] = idx;
        atomicAdd(&counts[g * NK + idx], 1u);
      }
    }
  }
}

// ---------------------------------------------------------------------------
// Kernel 3: dequantize -> out (N,C,T) + accumulate squared error for loss.
// Each thread handles 4 consecutive t: b128 streaming loads/stores.
// ---------------------------------------------------------------------------
__global__ void __launch_bounds__(256)
pq_dequant_kernel(const float* __restrict__ x,
                  const float* __restrict__ cb,
                  const unsigned* __restrict__ codeIdx,
                  float* __restrict__ out,
                  float* __restrict__ lossSum) {
  size_t o = ((size_t)blockIdx.x * blockDim.x + threadIdx.x) * 4;  // < NEL
  int t     = (int)(o & (T_L - 1));            // multiple of 4
  int c     = (int)((o >> 11) & (C_F - 1));
  int ni    = (int)(o >> 20);                  // C_F*T_L = 2^20
  int token = ni * T_L + t;
  int g     = c >> 7;
  int dg    = c & (DG - 1);

  unsigned i0 = codeIdx[(size_t)(token + 0) * NG + g];
  unsigned i1 = codeIdx[(size_t)(token + 1) * NG + g];
  unsigned i2 = codeIdx[(size_t)(token + 2) * NG + g];
  unsigned i3 = codeIdx[(size_t)(token + 3) * NG + g];

  float4 qv;
  qv.x = cb[((size_t)(g * NK + i0) << 7) + dg];
  qv.y = cb[((size_t)(g * NK + i1) << 7) + dg];
  qv.z = cb[((size_t)(g * NK + i2) << 7) + dg];
  qv.w = cb[((size_t)(g * NK + i3) << 7) + dg];

  const float4 xv = *(const float4*)(x + o);
  *(float4*)(out + o) = qv;                    // forward value of STE output

  float d, acc;
  d = qv.x - xv.x; acc  = d * d;
  d = qv.y - xv.y; acc += d * d;
  d = qv.z - xv.z; acc += d * d;
  d = qv.w - xv.w; acc += d * d;

  __shared__ float red[256];
  red[threadIdx.x] = acc;
  __syncthreads();
  #pragma unroll
  for (int s = 128; s > 0; s >>= 1) {
    if (threadIdx.x < s) red[threadIdx.x] += red[threadIdx.x + s];
    __syncthreads();
  }
  if (threadIdx.x == 0) atomicAdd(lossSum, red[0]);
}

// ---------------------------------------------------------------------------
// Kernel 4: perplexity from histogram + loss finalization. One block.
// ---------------------------------------------------------------------------
__global__ void pq_finalize_kernel(const unsigned* __restrict__ counts,
                                   const float* __restrict__ lossSum,
                                   float* __restrict__ outTail) {
  __shared__ float red[1024];
  __shared__ float perpAcc;
  int tid = threadIdx.x;
  if (tid == 0) perpAcc = 0.0f;
  __syncthreads();
  const float inv = 1.0f / (float)NTOK;
  for (int g = 0; g < NG; ++g) {
    float p = (float)counts[g * NK + tid] * inv;
    red[tid] = p * logf(p + 1e-10f);
    __syncthreads();
    for (int s = 512; s > 0; s >>= 1) {
      if (tid < s) red[tid] += red[tid + s];
      __syncthreads();
    }
    if (tid == 0) perpAcc += expf(-red[0]);
    __syncthreads();
  }
  if (tid == 0) {
    outTail[0] = (1.0f + BETA) * (*lossSum) / (float)NEL;  // loss
    outTail[1] = perpAcc / (float)NG;                      // perplexity
  }
}

// ---------------------------------------------------------------------------
extern "C" void kernel_launch(void* const* d_in, const int* in_sizes, int n_in,
                              void* d_out, int out_size, void* d_ws, size_t ws_size,
                              hipStream_t stream) {
  const float* x  = (const float*)d_in[0];   // (32, 512, 2048)
  const float* cb = (const float*)d_in[1];   // (4, 1024, 128)
  float* out = (float*)d_out;                // [NEL] out, [NEL]=loss, [NEL+1]=perp

  // Workspace layout (all 256B-aligned chunks):
  char* ws = (char*)d_ws;
  _Float16* cb16    = (_Float16*)ws;                                   // 1 MB
  float*    cnorm   = (float*)(ws + (1u << 20));                       // 16 KB
  unsigned* counts  = (unsigned*)(ws + (1u << 20) + (16u << 10));      // 16 KB
  float*    lossSum = (float*)(ws + (1u << 20) + (32u << 10));         // 256 B slot
  unsigned* codeIdx = (unsigned*)(ws + (1u << 20) + (32u << 10) + 256);// 1 MB

  // 1) codebook f16 + norms, zero accumulators
  pq_prep_kernel<<<(NG * NK + 255) / 256, 256, 0, stream>>>(
      cb, cb16, cnorm, counts, lossSum);

  // 2) fused WMMA distance + argmin: 1024 blocks x 8 waves (each wave: 32 tok)
  pq_argmin_kernel<<<(NTOK / 256) * NG, 256, 0, stream>>>(
      x, cb16, cnorm, codeIdx, counts);

  // 3) dequant + output + loss accumulation (4 elems/thread, b128 streams)
  pq_dequant_kernel<<<NEL / 4 / 256, 256, 0, stream>>>(
      x, cb, codeIdx, out, lossSum);

  // 4) scalars
  pq_finalize_kernel<<<1, 1024, 0, stream>>>(counts, lossSum, out + NEL);
}
